// EffGCNBlock_71554155151645
// MI455X (gfx1250) — compile-verified, matrix operand order checked by reading, not run on hardware
//
#include <hip/hip_runtime.h>
#include <hip/hip_bf16.h>
#include <math.h>

// ---------------- problem constants ----------------
constexpr int Nn = 128, Cc = 64, Tt = 256, Vv = 25, Kp = 3, Ee = 8, Rr = 16;
constexpr float EPS_BN = 1e-5f;

// GCN kernel tiling
constexpr int T_TILE = 8;           // t's per block
constexpr int PPOS   = T_TILE * Vv; // 200 real positions
constexpr int NTN    = 13;          // real N-tiles of 16 (208 positions)
constexpr int PPAD   = 224;         // padded rows so tile pairs (nt,nt+1) never read OOB
constexpr int KTOT   = 5 * Cc;      // 320 stacked K (3 phys + learned + dyn)
constexpr int KS_N   = KTOT / 32;   // 10 WMMA k-steps
constexpr int SAGG   = 328;         // LDS row stride (bf16) for K dim, padded

// TCN kernel tiling
constexpr int T_TILE2 = 32;
constexpr int TZ      = T_TILE2 + 8;   // halo 4 each side
constexpr int P2      = T_TILE2 * Vv;  // 800 = 50 N-tiles of 16
constexpr int SU      = 72;            // LDS row stride (bf16) for K=64, padded

// ---------------- workspace layout (bytes) ----------------
constexpr size_t WS_AL    = 0;          //   625 f32   normalized learned adjacency
constexpr size_t WS_WPACK = 2560;       // 20480 bf16  subset_w packed [o][b*64+c]
constexpr size_t WS_PWP   = 43520;      //  4096 bf16  pw_w packed [o][c]
constexpr size_t WS_XPOOL = 51712;      // N*C*V f32
constexpr size_t WS_ADYN  = 870912;     // N*V*V f32
constexpr size_t WS_YT    = 1190912;    // N*C*T f32
constexpr size_t WS_YV    = 9579520;    // N*C*V f32
constexpr size_t WS_TATT  = 10398720;   // N*C*T f32
constexpr size_t WS_VATT  = 18787328;   // N*C*V f32
constexpr size_t WS_Y     = 19606528;   // N*C*T*V f32 (gcn output, post BN+ReLU)
// total ~229.3 MB

typedef __attribute__((ext_vector_type(16))) __bf16 v16bf;
typedef __attribute__((ext_vector_type(8)))  __bf16 v8bf;
typedef __attribute__((ext_vector_type(8)))  float  v8f;

// Fragment load per CDNA5 16-bit A/B layout: lane L (r=L&15, h=L>>4) holds
// elements e<8 -> k = k0+8h+e, e>=8 -> k = k0+16+8h+(e-8). With rows stored
// [row][k] (k contiguous, 16B aligned) this is two ds_load_b128 per lane.
static __device__ inline v16bf load_frag(const __bf16* p) {
    v8bf lo = *(const v8bf*)(p);
    v8bf hi = *(const v8bf*)(p + 16);
    v16bf f;
#pragma unroll
    for (int i = 0; i < 8; ++i) { f[i] = lo[i]; f[i + 8] = hi[i]; }
    return f;
}

static __device__ inline float bn1f(float x, const float* bn, int c) {
    float g = bn[c], b = bn[Cc + c], m = bn[2 * Cc + c], v = bn[3 * Cc + c];
    return (x - m) * (g * rsqrtf(v + EPS_BN)) + b;
}

// ---------------- kernel 1: prep (learned adjacency norm + weight packing) ----------------
__global__ void prep_kernel(const float* __restrict__ A_learned,
                            const float* __restrict__ subset_w,
                            const float* __restrict__ pw_w,
                            char* __restrict__ ws) {
    float*  Al = (float*)(ws + WS_AL);
    __bf16* Wp = (__bf16*)(ws + WS_WPACK);
    __bf16* Pw = (__bf16*)(ws + WS_PWP);
    __shared__ float d[32];
    const int tid = threadIdx.x;
    if (tid < Vv) {
        float s = 0.f;
        for (int j = 0; j < Vv; ++j) s += fabsf(A_learned[tid * Vv + j]);
        d[tid] = rsqrtf(fmaxf(s, 1e-6f));
    }
    __syncthreads();
    for (int i = tid; i < Vv * Vv; i += blockDim.x) {
        int r = i / Vv, c = i % Vv;
        Al[i] = d[r] * fabsf(A_learned[i]) * d[c];
    }
    for (int i = tid; i < Cc * KTOT; i += blockDim.x) {
        int o = i / KTOT, kb = i % KTOT, b = kb >> 6, c = kb & 63;
        Wp[i] = (__bf16)subset_w[(b * Cc + o) * Cc + c];
    }
    for (int i = tid; i < Cc * Cc; i += blockDim.x) Pw[i] = (__bf16)pw_w[i];
}

// ---------------- kernel 2: temporal mean pool of x ----------------
__global__ void pool_x_kernel(const float* __restrict__ x, char* __restrict__ ws) {
    float* xp = (float*)(ws + WS_XPOOL);
    int idx = blockIdx.x * blockDim.x + threadIdx.x; // (n*C + c)*V + v
    if (idx >= Nn * Cc * Vv) return;
    int v = idx % Vv, nc = idx / Vv;
    const float* p = x + (size_t)nc * Tt * Vv + v;
    float s = 0.f;
    for (int t = 0; t < Tt; ++t) s += p[t * Vv];
    xp[idx] = s * (1.0f / Tt);
}

// ---------------- kernel 3: dynamic adjacency (per sample) ----------------
__global__ void dyn_adj_kernel(const float* __restrict__ node_proj_w, char* __restrict__ ws) {
    const int n = blockIdx.x, tid = threadIdx.x;
    const float* xp = (float*)(ws + WS_XPOOL) + (size_t)n * Cc * Vv;
    float* Adyn = (float*)(ws + WS_ADYN) + (size_t)n * Vv * Vv;
    __shared__ float s_xp[Cc * Vv];
    __shared__ float s_emb[Vv][Ee];
    __shared__ float s_S[Vv][Vv];
    for (int i = tid; i < Cc * Vv; i += blockDim.x) s_xp[i] = xp[i];
    __syncthreads();
    if (tid < Vv * Ee) {
        int v = tid / Ee, e = tid % Ee;
        float s = 0.f;
        for (int c = 0; c < Cc; ++c) s += s_xp[c * Vv + v] * node_proj_w[e * Cc + c];
        s_emb[v][e] = s;
    }
    __syncthreads();
    if (tid < Vv) {
        float nn = 0.f;
        for (int e = 0; e < Ee; ++e) nn += s_emb[tid][e] * s_emb[tid][e];
        float inv = 1.0f / fmaxf(sqrtf(nn), 1e-12f);
        for (int e = 0; e < Ee; ++e) s_emb[tid][e] *= inv;
    }
    __syncthreads();
    for (int i = tid; i < Vv * Vv; i += blockDim.x) {
        int v = i / Vv, w = i % Vv;
        float s = 0.f;
        for (int e = 0; e < Ee; ++e) s += s_emb[v][e] * s_emb[w][e];
        s_S[v][w] = s;
    }
    __syncthreads();
    if (tid < Vv) {
        float mx = -1e30f;
        for (int w = 0; w < Vv; ++w) mx = fmaxf(mx, s_S[tid][w]);
        float ex[Vv], se = 0.f;
        for (int w = 0; w < Vv; ++w) { ex[w] = expf(s_S[tid][w] - mx); se += ex[w]; }
        float inv = 1.0f / se;
        for (int w = 0; w < Vv; ++w) Adyn[tid * Vv + w] = ex[w] * inv;
    }
}

// ---------------- kernel 4: fused GCN (5 aggregations + stacked-K WMMA GEMM + BN + ReLU) ----------------
__global__ __launch_bounds__(256) void gcn_kernel(const float* __restrict__ x,
                                                  const float* __restrict__ A_physical,
                                                  const float* __restrict__ bn_gcn,
                                                  char* __restrict__ ws) {
    extern __shared__ char smem[];
    float (*s_adj)[Vv][26] = (float (*)[Vv][26])smem;          // 5*25*26*4 = 13000 B
    __bf16* s_w   = (__bf16*)(smem + 13056);                   // 64*328*2  = 41984 B
    __bf16* s_agg = (__bf16*)(smem + 13056 + 41984);           // 224*328*2 = 146944 B
    const int n = blockIdx.x, t0 = blockIdx.y * T_TILE, tid = threadIdx.x;
    const float*  Al   = (const float*)(ws + WS_AL);
    const float*  Adyn = (const float*)(ws + WS_ADYN) + (size_t)n * Vv * Vv;
    const __bf16* Wp   = (const __bf16*)(ws + WS_WPACK);
    float*        y    = (float*)(ws + WS_Y);

    // 5 adjacency matrices -> LDS
    for (int i = tid; i < 5 * Vv * Vv; i += 256) {
        int b = i / (Vv * Vv), rem = i % (Vv * Vv), r = rem / Vv, j = rem % Vv;
        float a;
        if (b < Kp)      a = A_physical[i];
        else if (b == 3) a = Al[rem];
        else             a = Adyn[rem];
        s_adj[b][r][j] = a;
    }
    // packed weights -> LDS (row stride SAGG)
    for (int i = tid; i < Cc * KTOT; i += 256)
        s_w[(i / KTOT) * SAGG + (i % KTOT)] = Wp[i];
    // zero tail rows of agg (positions 200..223 incl. pair-padding tile)
    for (int i = tid; i < (PPAD - PPOS) * SAGG; i += 256)
        s_agg[PPOS * SAGG + i] = (__bf16)0.0f;
    __syncthreads();

    // Aggregation: agg[b*64+c, tt*25+j] = sum_i x[n,c,t0+tt,i] * adj_b[i,j]  (fp32 VALU)
    for (int rix = tid; rix < Cc * T_TILE; rix += 256) {
        int c = rix >> 3, tt = rix & 7;
        const float* xr = x + (((size_t)n * Cc + c) * Tt + (t0 + tt)) * Vv;
        float xv[Vv];
#pragma unroll
        for (int i = 0; i < Vv; ++i) xv[i] = xr[i];
#pragma unroll 1
        for (int b = 0; b < 5; ++b) {
#pragma unroll 1
            for (int j = 0; j < Vv; ++j) {
                float s = 0.f;
#pragma unroll
                for (int i = 0; i < Vv; ++i) s += xv[i] * s_adj[b][i][j];
                s_agg[(tt * Vv + j) * SAGG + b * Cc + c] = (__bf16)s;
            }
        }
    }
    __syncthreads();

    // GEMM: out[64 x 208] = W[64 x 320] * aggT[320 x 224] via v_wmma_f32_16x16x32_bf16.
    // mt is wave-invariant -> A fragments (10 x v16bf) hoisted and reused.
    // Two independent N-tiles per wave iteration: two live B fragments + two
    // accumulator chains prevent register coalescing, keeping next-step loads in
    // flight across each WMMA (s_wait_dscnt > 0) and doubling MMA per loop trip.
    const int wave = tid >> 5, lane = tid & 31, r = lane & 15, half = lane >> 4;
    const int mt = wave & 3;
    const __bf16* arow = s_w + (mt * 16 + r) * SAGG + 8 * half;
    v16bf afr[KS_N];
#pragma unroll
    for (int ks = 0; ks < KS_N; ++ks) afr[ks] = load_frag(arow + ks * 32);

    auto epilogue = [&](const v8f& acc, int nt) {
        const int p = nt * 16 + r;
        if (p < PPOS) {
            const int tt = p / Vv, j = p % Vv;
#pragma unroll
            for (int i = 0; i < 8; ++i) {
                int o = mt * 16 + 8 * half + i;    // C/D layout: m = vgpr + 8*(lane>>4)
                float val = fmaxf(bn1f(acc[i], bn_gcn, o), 0.0f);
                y[(((size_t)n * Cc + o) * Tt + (t0 + tt)) * Vv + j] = val;
            }
        }
    };

    for (int ntb = (wave >> 2) * 2; ntb < NTN; ntb += 4) {
        const __bf16* brow0 = s_agg + (ntb * 16 + r) * SAGG + 8 * half;
        const __bf16* brow1 = brow0 + 16 * SAGG;   // nt = ntb+1 (zeros if == NTN)
        v8f acc0 = {0.f, 0.f, 0.f, 0.f, 0.f, 0.f, 0.f, 0.f};
        v8f acc1 = {0.f, 0.f, 0.f, 0.f, 0.f, 0.f, 0.f, 0.f};
        v16bf b0 = load_frag(brow0);
        v16bf b1 = load_frag(brow1);
#pragma unroll
        for (int ks = 0; ks < KS_N; ++ks) {
            v16bf nb0 = b0, nb1 = b1;
            if (ks + 1 < KS_N) {
                nb0 = load_frag(brow0 + (ks + 1) * 32);
                nb1 = load_frag(brow1 + (ks + 1) * 32);
            }
            acc0 = __builtin_amdgcn_wmma_f32_16x16x32_bf16(false, afr[ks], false, b0,
                                                           (short)0, acc0, false, false);
            acc1 = __builtin_amdgcn_wmma_f32_16x16x32_bf16(false, afr[ks], false, b1,
                                                           (short)0, acc1, false, false);
            b0 = nb0; b1 = nb1;
        }
        epilogue(acc0, ntb);
        epilogue(acc1, ntb + 1);
    }
}

// ---------------- kernel 5: pools of y for attention ----------------
__global__ void pool_y_kernel(char* __restrict__ ws) {
    const float* y = (const float*)(ws + WS_Y);
    float* yt = (float*)(ws + WS_YT);
    float* yv = (float*)(ws + WS_YV);
    const int nc = blockIdx.x, t = threadIdx.x; // 256 threads == T
    __shared__ float s_v[Vv];
    if (t < Vv) s_v[t] = 0.0f;
    __syncthreads();
    const float* p = y + (size_t)nc * Tt * Vv + (size_t)t * Vv;
    float s = 0.f, vals[Vv];
#pragma unroll
    for (int j = 0; j < Vv; ++j) { vals[j] = p[j]; s += vals[j]; }
    yt[(size_t)nc * Tt + t] = s * (1.0f / Vv);
#pragma unroll
    for (int j = 0; j < Vv; ++j) atomicAdd(&s_v[j], vals[j]);
    __syncthreads();
    if (t < Vv) yv[(size_t)nc * Vv + t] = s_v[t] * (1.0f / Tt);
}

// ---------------- kernel 6: attention maps ----------------
__global__ void att_kernel(const float* __restrict__ att_fcn_w, const float* __restrict__ att_fcn_bn,
                           const float* __restrict__ att_t_w,  const float* __restrict__ att_t_b,
                           const float* __restrict__ att_v_w,  const float* __restrict__ att_v_b,
                           char* __restrict__ ws) {
    const int n = blockIdx.x, tid = threadIdx.x;
    const float* yt = (const float*)(ws + WS_YT) + (size_t)n * Cc * Tt;
    const float* yv = (const float*)(ws + WS_YV) + (size_t)n * Cc * Vv;
    float* tatt = (float*)(ws + WS_TATT) + (size_t)n * Cc * Tt;
    float* vatt = (float*)(ws + WS_VATT) + (size_t)n * Cc * Vv;
    __shared__ float s_h[Rr][Tt + Vv + 7];
    for (int s = tid; s < Tt + Vv; s += 256) {
        float acc[Rr];
#pragma unroll
        for (int r = 0; r < Rr; ++r) acc[r] = 0.f;
        for (int c = 0; c < Cc; ++c) {
            float xv = (s < Tt) ? yt[c * Tt + s] : yv[c * Vv + (s - Tt)];
#pragma unroll
            for (int r = 0; r < Rr; ++r) acc[r] += att_fcn_w[r * Cc + c] * xv;
        }
#pragma unroll
        for (int r = 0; r < Rr; ++r) {
            float g = att_fcn_bn[r], b = att_fcn_bn[Rr + r];
            float m = att_fcn_bn[2 * Rr + r], va = att_fcn_bn[3 * Rr + r];
            float h = (acc[r] - m) * (g * rsqrtf(va + EPS_BN)) + b;
            h = h * fminf(fmaxf(h + 3.0f, 0.0f), 6.0f) * (1.0f / 6.0f); // hswish
            s_h[r][s] = h;
        }
    }
    __syncthreads();
    for (int i = tid; i < Cc * Tt; i += 256) {
        int c = i / Tt, t = i % Tt;
        float s = att_t_b[c];
#pragma unroll
        for (int r = 0; r < Rr; ++r) s += att_t_w[c * Rr + r] * s_h[r][t];
        tatt[i] = 1.0f / (1.0f + expf(-s));
    }
    for (int i = tid; i < Cc * Vv; i += 256) {
        int c = i / Vv, v = i % Vv;
        float s = att_v_b[c];
#pragma unroll
        for (int r = 0; r < Rr; ++r) s += att_v_w[c * Rr + r] * s_h[r][Tt + v];
        vatt[i] = 1.0f / (1.0f + expf(-s));
    }
}

// ---------------- kernel 7: fused TCN (att * BN1+ReLU + dwconv9 + pointwise WMMA + BN2 + res + ReLU) ----------------
__global__ __launch_bounds__(256) void tcn_kernel(const float* __restrict__ x,
                                                  const float* __restrict__ dw_w,
                                                  const float* __restrict__ bn_tcn1,
                                                  const float* __restrict__ bn_tcn2,
                                                  char* __restrict__ ws,
                                                  float* __restrict__ out) {
    extern __shared__ char smem[];
    float*  s_dw = (float*)smem;                                  // 64*9*4  = 2304 B
    __bf16* s_z  = (__bf16*)(smem + 2304);                        // 64*40*26*2 = 133120 B
    __bf16* s_u  = (__bf16*)(smem + 2304 + 133120);               // 800*72*2   = 115200 B
    __bf16* s_pw = (__bf16*)(smem + 2304 + 133120 + 115200);      // 64*72*2    = 9216 B
    const int n = blockIdx.x, t0 = blockIdx.y * T_TILE2, tid = threadIdx.x;
    const float* yb   = (const float*)(ws + WS_Y)    + (size_t)n * Cc * Tt * Vv;
    const float* tatt = (const float*)(ws + WS_TATT) + (size_t)n * Cc * Tt;
    const float* vatt = (const float*)(ws + WS_VATT) + (size_t)n * Cc * Vv;
    const __bf16* Pw  = (const __bf16*)(ws + WS_PWP);

    for (int i = tid; i < Cc * 9; i += 256) s_dw[i] = dw_w[i];
    for (int i = tid; i < Cc * Cc; i += 256) s_pw[(i >> 6) * SU + (i & 63)] = Pw[i];

    // z tile with halo: z = relu(bn1(y * t_att * v_att))
    for (int rix = tid; rix < Cc * TZ; rix += 256) {
        int c = rix / TZ, tl = rix % TZ;
        int t = t0 + tl - 4;
        __bf16* zr = s_z + (c * TZ + tl) * 26;
        if (t >= 0 && t < Tt) {
            float g = bn_tcn1[c], be = bn_tcn1[Cc + c];
            float m = bn_tcn1[2 * Cc + c], va = bn_tcn1[3 * Cc + c];
            float sc = g * rsqrtf(va + EPS_BN);
            float ta = tatt[c * Tt + t];
            const float* yr = yb + ((size_t)c * Tt + t) * Vv;
#pragma unroll
            for (int v = 0; v < Vv; ++v) {
                float z = yr[v] * ta * vatt[c * Vv + v];
                zr[v] = (__bf16)fmaxf((z - m) * sc + be, 0.0f);
            }
        } else {
#pragma unroll
            for (int v = 0; v < Vv; ++v) zr[v] = (__bf16)0.0f;
        }
    }
    __syncthreads();

    // depthwise 9-tap over T -> transposed [pos][c] for WMMA B
    for (int rix = tid; rix < Cc * T_TILE2; rix += 256) {
        int c = rix >> 5, tt = rix & 31;
        float w9[9];
#pragma unroll
        for (int j = 0; j < 9; ++j) w9[j] = s_dw[c * 9 + j];
#pragma unroll 1
        for (int v = 0; v < Vv; ++v) {
            float s = 0.f;
#pragma unroll
            for (int j = 0; j < 9; ++j) s += w9[j] * (float)s_z[(c * TZ + tt + j) * 26 + v];
            s_u[(tt * Vv + v) * SU + c] = (__bf16)s;
        }
    }
    __syncthreads();

    // pointwise GEMM 64 x 800 x 64 via WMMA, fused BN2 + residual + ReLU.
    // mt wave-invariant -> both A fragments hoisted; two N-tiles per iteration
    // give 4 independent B fragments / 2 accumulators, keeping LDS loads in
    // flight across the WMMAs.
    const int wave = tid >> 5, lane = tid & 31, r = lane & 15, half = lane >> 4;
    const int mt = wave & 3;
    const __bf16* arow = s_pw + (mt * 16 + r) * SU + 8 * half;
    v16bf a0 = load_frag(arow);
    v16bf a1 = load_frag(arow + 32);

    auto epilogue = [&](const v8f& acc, int nt) {
        const int p = nt * 16 + r, tt = p / Vv, v = p % Vv;
#pragma unroll
        for (int i = 0; i < 8; ++i) {
            int o = mt * 16 + 8 * half + i;
            float val = bn1f(acc[i], bn_tcn2, o);
            size_t gi = (((size_t)n * Cc + o) * Tt + (t0 + tt)) * Vv + v;
            out[gi] = fmaxf(val + x[gi], 0.0f);
        }
    };

    for (int ntb = (wave >> 2) * 2; ntb < P2 / 16; ntb += 4) {
        const __bf16* brow0 = s_u + (ntb * 16 + r) * SU + 8 * half;
        const __bf16* brow1 = brow0 + 16 * SU;
        v16bf b00 = load_frag(brow0);
        v16bf b10 = load_frag(brow1);
        v16bf b01 = load_frag(brow0 + 32);
        v16bf b11 = load_frag(brow1 + 32);
        v8f acc0 = {0.f, 0.f, 0.f, 0.f, 0.f, 0.f, 0.f, 0.f};
        v8f acc1 = {0.f, 0.f, 0.f, 0.f, 0.f, 0.f, 0.f, 0.f};
        acc0 = __builtin_amdgcn_wmma_f32_16x16x32_bf16(false, a0, false, b00,
                                                       (short)0, acc0, false, false);
        acc1 = __builtin_amdgcn_wmma_f32_16x16x32_bf16(false, a0, false, b10,
                                                       (short)0, acc1, false, false);
        acc0 = __builtin_amdgcn_wmma_f32_16x16x32_bf16(false, a1, false, b01,
                                                       (short)0, acc0, false, false);
        acc1 = __builtin_amdgcn_wmma_f32_16x16x32_bf16(false, a1, false, b11,
                                                       (short)0, acc1, false, false);
        epilogue(acc0, ntb);
        epilogue(acc1, ntb + 1);
    }
}

// ---------------- launcher ----------------
extern "C" void kernel_launch(void* const* d_in, const int* in_sizes, int n_in,
                              void* d_out, int out_size, void* d_ws, size_t ws_size,
                              hipStream_t stream) {
    (void)in_sizes; (void)n_in; (void)out_size; (void)ws_size;
    const float* x          = (const float*)d_in[0];
    const float* A_physical = (const float*)d_in[1];
    const float* A_learned  = (const float*)d_in[2];
    const float* node_proj  = (const float*)d_in[3];
    const float* subset_w   = (const float*)d_in[4];
    const float* bn_gcn     = (const float*)d_in[5];
    const float* att_fcn_w  = (const float*)d_in[6];
    const float* att_fcn_bn = (const float*)d_in[7];
    const float* att_t_w    = (const float*)d_in[8];
    const float* att_t_b    = (const float*)d_in[9];
    const float* att_v_w    = (const float*)d_in[10];
    const float* att_v_b    = (const float*)d_in[11];
    const float* dw_w       = (const float*)d_in[12];
    const float* pw_w       = (const float*)d_in[13];
    const float* bn_tcn1    = (const float*)d_in[14];
    const float* bn_tcn2    = (const float*)d_in[15];
    char* ws = (char*)d_ws;
    float* out = (float*)d_out;

    prep_kernel<<<1, 256, 0, stream>>>(A_learned, subset_w, pw_w, ws);
    pool_x_kernel<<<(Nn * Cc * Vv + 255) / 256, 256, 0, stream>>>(x, ws);
    dyn_adj_kernel<<<Nn, 256, 0, stream>>>(node_proj, ws);

    const size_t gcn_lds = 13056 + 41984 + (size_t)PPAD * SAGG * 2;   // 201,984 B
    gcn_kernel<<<dim3(Nn, Tt / T_TILE), 256, gcn_lds, stream>>>(x, A_physical, bn_gcn, ws);

    pool_y_kernel<<<Nn * Cc, 256, 0, stream>>>(ws);
    att_kernel<<<Nn, 256, 0, stream>>>(att_fcn_w, att_fcn_bn, att_t_w, att_t_b,
                                       att_v_w, att_v_b, ws);

    const size_t tcn_lds = 2304 + 133120 + 115200 + 9216;             // 259,840 B
    tcn_kernel<<<dim3(Nn, Tt / T_TILE2), 256, tcn_lds, stream>>>(x, dw_w, bn_tcn1,
                                                                 bn_tcn2, ws, out);
}
